// MultiQueryAttention_20229295964509
// MI455X (gfx1250) — matrix operand order repaired and checked by reading, not compiled, not run on hardware
//
#include <hip/hip_runtime.h>

#define B_    8
#define T_    1024
#define C_    2048
#define H_    16
#define KVH_  4
#define D_    128
#define NKV_  1024   // KV projection output features

typedef _Float16 h16;
typedef __attribute__((ext_vector_type(16))) _Float16 v16h;
typedef __attribute__((ext_vector_type(8)))  _Float16 v8h;
typedef __attribute__((ext_vector_type(4)))  _Float16 v4h;
typedef __attribute__((ext_vector_type(8)))  float    v8f;

static __device__ __forceinline__ v8f wmma_f16(v16h a, v16h b, v8f c) {
  // D = A(16x32 f16) * B(32x16 f16) + C(16x16 f32)
  return __builtin_amdgcn_wmma_f32_16x16x32_f16(false, a, false, b, (short)0, c, false, false);
}

// ---- A fragment (16x32 f16): rows m0..m0+15, cols k0..k0+31 of row-major [.., ld].
// ISA layout: lanes 0-15 = rows with K 8h..8h+7 in VGPR0-3 and K 16+8h.. in VGPR4-7
// => per lane: two contiguous 16B chunks at k0+8*half and k0+16+8*half.
static __device__ __forceinline__ v16h load_a_frag(const h16* src, int ld, int m0, int k0, int lane) {
  const int m = m0 + (lane & 15);
  const int half = lane >> 4;
  const h16* p = src + (size_t)m * ld + k0 + 8 * half;
  v8h lo = *(const v8h*)(p);
  v8h hi = *(const v8h*)(p + 16);
  v16h r;
#pragma unroll
  for (int j = 0; j < 8; ++j) { r[j] = lo[j]; r[j + 8] = hi[j]; }
  return r;
}

// ---- B fragment (32x16 f16): B[K=k][N=n] = W[n0+n][k0+k] with W row-major [.., ld].
// ISA layout: lane = column n, K 8h..8h+7 then K 16+8h..16+8h+7 (matches A)
// => per lane: two contiguous 16B chunks at k0+8*half and k0+16+8*half.
static __device__ __forceinline__ v16h load_b_frag(const h16* w, int ld, int n0, int k0, int lane) {
  const int n = n0 + (lane & 15);
  const int half = lane >> 4;
  const h16* p = w + (size_t)n * ld + k0 + 8 * half;
  v8h lo = *(const v8h*)(p);
  v8h hi = *(const v8h*)(p + 16);
  v16h r;
#pragma unroll
  for (int j = 0; j < 8; ++j) { r[j] = lo[j]; r[j + 8] = hi[j]; }
  return r;
}

// ---- 16(M) x 64(N) GEMM tile per wave: D = A[M,K] * W[N,K]^T
// Ping-pong double buffering, 2x K-unroll: each half issues the other buffer's
// 10 b128 loads, then runs its 4 WMMAs. Loop-carried fragments are defined
// directly by loads (no register rotation -> no v_mov copy block).
static __device__ __forceinline__ void gemm_core(const h16* __restrict__ A, const h16* __restrict__ W,
                                                 int K, int m0, int n0, int lane, v8f* acc) {
  v16h a0 = load_a_frag(A, K, m0, 0, lane);
  v16h p0 = load_b_frag(W, K, n0 +  0, 0, lane);
  v16h p1 = load_b_frag(W, K, n0 + 16, 0, lane);
  v16h p2 = load_b_frag(W, K, n0 + 32, 0, lane);
  v16h p3 = load_b_frag(W, K, n0 + 48, 0, lane);
#pragma unroll 1
  for (int k0 = 0; k0 < K; k0 += 64) {
    // prefetch second half (k0+32) into buffer 1
    v16h a1 = load_a_frag(A, K, m0, k0 + 32, lane);
    v16h q0 = load_b_frag(W, K, n0 +  0, k0 + 32, lane);
    v16h q1 = load_b_frag(W, K, n0 + 16, k0 + 32, lane);
    v16h q2 = load_b_frag(W, K, n0 + 32, k0 + 32, lane);
    v16h q3 = load_b_frag(W, K, n0 + 48, k0 + 32, lane);
    acc[0] = wmma_f16(a0, p0, acc[0]);
    acc[1] = wmma_f16(a0, p1, acc[1]);
    acc[2] = wmma_f16(a0, p2, acc[2]);
    acc[3] = wmma_f16(a0, p3, acc[3]);
    // reload buffer 0 for k0+64 (wrap keeps the tail loads in-bounds)
    const int kn = (k0 + 64 < K) ? (k0 + 64) : 0;
    a0 = load_a_frag(A, K, m0, kn, lane);
    p0 = load_b_frag(W, K, n0 +  0, kn, lane);
    p1 = load_b_frag(W, K, n0 + 16, kn, lane);
    p2 = load_b_frag(W, K, n0 + 32, kn, lane);
    p3 = load_b_frag(W, K, n0 + 48, kn, lane);
    acc[0] = wmma_f16(a1, q0, acc[0]);
    acc[1] = wmma_f16(a1, q1, acc[1]);
    acc[2] = wmma_f16(a1, q2, acc[2]);
    acc[3] = wmma_f16(a1, q3, acc[3]);
  }
}

// ---------------- f32 -> f16 convert (b128 loads, 4 elems/thread) ----------------
__global__ __launch_bounds__(256) void cvt_kernel(const float4* __restrict__ s, v4h* __restrict__ d, int n4) {
  for (int i = blockIdx.x * blockDim.x + threadIdx.x; i < n4; i += blockDim.x * gridDim.x) {
    float4 v = s[i];
    v4h o;
    o[0] = (h16)v.x; o[1] = (h16)v.y; o[2] = (h16)v.z; o[3] = (h16)v.w;
    d[i] = o;
  }
}

// ---------------- Q projection: Qb[m, n] = x @ Wq^T + bq (f16 out) ----------------
__global__ __launch_bounds__(256) void gemm_q_kernel(const h16* __restrict__ xh, const h16* __restrict__ Wh,
                                                     const float* __restrict__ bias, h16* __restrict__ Qb) {
  const int lane = threadIdx.x & 31;
  const int gw = (blockIdx.x * blockDim.x + threadIdx.x) >> 5;
  const int ntile = C_ / 64;
  const int m0 = (gw / ntile) * 16;
  const int n0 = (gw % ntile) * 64;
  v8f acc[4];
#pragma unroll
  for (int t = 0; t < 4; ++t)
#pragma unroll
    for (int r = 0; r < 8; ++r) acc[t][r] = 0.f;
  gemm_core(xh, Wh, C_, m0, n0, lane, acc);
  const int half = lane >> 4, nn = lane & 15;
#pragma unroll
  for (int t = 0; t < 4; ++t) {
    const int n = n0 + 16 * t + nn;
    const float bn = bias[n];
#pragma unroll
    for (int r = 0; r < 8; ++r) {
      const int m = m0 + r + 8 * half;
      Qb[(size_t)m * C_ + n] = (h16)(acc[t][r] + bn);
    }
  }
}

// ---------------- KV projection with scatter: K -> [b,kvh,t,d], V -> transposed [b,kvh,d,t] ----------------
__global__ __launch_bounds__(256) void gemm_kv_kernel(const h16* __restrict__ xh, const h16* __restrict__ Wh,
                                                      const float* __restrict__ bias,
                                                      h16* __restrict__ Kb, h16* __restrict__ Vtb) {
  const int lane = threadIdx.x & 31;
  const int gw = (blockIdx.x * blockDim.x + threadIdx.x) >> 5;
  const int ntile = NKV_ / 64;
  const int m0 = (gw / ntile) * 16;
  const int n0 = (gw % ntile) * 64;
  v8f acc[4];
#pragma unroll
  for (int t = 0; t < 4; ++t)
#pragma unroll
    for (int r = 0; r < 8; ++r) acc[t][r] = 0.f;
  gemm_core(xh, Wh, C_, m0, n0, lane, acc);
  const int half = lane >> 4, nn = lane & 15;
#pragma unroll
  for (int t = 0; t < 4; ++t) {
    const int n = n0 + 16 * t + nn;           // feature: kvh*256 + which*128 + d
    const float bn = bias[n];
    const int kvh = n >> 8;
    const int which = (n >> 7) & 1;
    const int d = n & 127;
#pragma unroll
    for (int r = 0; r < 8; ++r) {
      const int row = m0 + r + 8 * half;      // b*T + t
      const int b = row >> 10;
      const int tt = row & (T_ - 1);
      const float v = acc[t][r] + bn;
      if (which == 0)
        Kb[(((size_t)b * KVH_ + kvh) * T_ + tt) * D_ + d] = (h16)v;
      else
        Vtb[(((size_t)b * KVH_ + kvh) * D_ + d) * T_ + tt] = (h16)v;
    }
  }
}

// ---------------- Flash attention: 1 wave handles a 16-query tile of one (b,h) ----------------
__global__ __launch_bounds__(32) void attn_kernel(const h16* __restrict__ Qb, const h16* __restrict__ Kb,
                                                  const h16* __restrict__ Vtb, h16* __restrict__ Yb) {
  __shared__ __align__(32) h16 lds_p[16 * 32];
  const int lane = threadIdx.x & 31;
  const int gw = blockIdx.x;                  // B*H*(T/16) waves
  const int qt = gw & (T_ / 16 - 1);
  const int bh = gw >> 6;
  const int h = bh & (H_ - 1);
  const int b = bh >> 4;
  const int kvh = h >> 2;                     // GQA: 4 query heads per kv head
  const int q0 = qt * 16;
  const int half = lane >> 4;
  const int nn = lane & 15;

  const h16* Qh = Qb + (size_t)b * T_ * C_ + h * D_;                 // rows t, ld C_
  const h16* Kh = Kb + ((size_t)b * KVH_ + kvh) * (size_t)(T_ * D_); // [T, D]
  const h16* Vh = Vtb + ((size_t)b * KVH_ + kvh) * (size_t)(D_ * T_);// [D, T] (transposed V)

  v16h qf[4];
#pragma unroll
  for (int c = 0; c < 4; ++c) qf[c] = load_a_frag(Qh, C_, q0, c * 32, lane);

  float mrow[8], lrow[8];
  v8f acc[8];                                 // Y: 16 x 128 f32
#pragma unroll
  for (int r = 0; r < 8; ++r) { mrow[r] = -1e30f; lrow[r] = 0.f; }
#pragma unroll
  for (int t = 0; t < 8; ++t)
#pragma unroll
    for (int r = 0; r < 8; ++r) acc[t][r] = 0.f;

  const float scale = 0.08838834764831845f;   // 1/sqrt(128)

  for (int n0 = 0; n0 <= q0 + 15; n0 += 32) { // causal key blocks of 32
    // ---- issue all 8 K fragments (16 b128 loads), then the 8 S WMMAs
    v16h kf[8];
#pragma unroll
    for (int c = 0; c < 4; ++c) {
      kf[2 * c]     = load_b_frag(Kh, D_, n0,      c * 32, lane);
      kf[2 * c + 1] = load_b_frag(Kh, D_, n0 + 16, c * 32, lane);
    }
    v8f s0, s1;
#pragma unroll
    for (int r = 0; r < 8; ++r) { s0[r] = 0.f; s1[r] = 0.f; }
#pragma unroll
    for (int c = 0; c < 4; ++c) {
      s0 = wmma_f16(qf[c], kf[2 * c], s0);
      s1 = wmma_f16(qf[c], kf[2 * c + 1], s1);
    }

    // ---- issue V fragments now: independent of softmax, loads overlap the exp/reduce math
    v16h vf[8];
#pragma unroll
    for (int t = 0; t < 8; ++t) vf[t] = load_b_frag(Vh, T_, t * 16, n0, lane);

    // scale + causal mask (D-layout: row = r + 8*half, col = nn)
#pragma unroll
    for (int r = 0; r < 8; ++r) {
      const int q = q0 + r + 8 * half;
      float a0 = s0[r] * scale;
      float a1 = s1[r] * scale;
      if (n0 + nn > q) a0 = -1e30f;
      if (n0 + 16 + nn > q) a1 = -1e30f;
      s0[r] = a0; s1[r] = a1;
    }
    // row max across the 16 lanes of each half
    float tm[8];
#pragma unroll
    for (int r = 0; r < 8; ++r) tm[r] = fmaxf(s0[r], s1[r]);
#pragma unroll
    for (int i = 0; i < 4; ++i) {
      const int off = 1 << i;
#pragma unroll
      for (int r = 0; r < 8; ++r) tm[r] = fmaxf(tm[r], __shfl_xor(tm[r], off, 32));
    }
    float alpha[8];
#pragma unroll
    for (int r = 0; r < 8; ++r) {
      const float mnew = fmaxf(mrow[r], tm[r]);
      alpha[r] = __expf(mrow[r] - mnew);
      mrow[r] = mnew;
    }
#pragma unroll
    for (int r = 0; r < 8; ++r) {
      s0[r] = __expf(s0[r] - mrow[r]);
      s1[r] = __expf(s1[r] - mrow[r]);
    }
    float ts[8];
#pragma unroll
    for (int r = 0; r < 8; ++r) ts[r] = s0[r] + s1[r];
#pragma unroll
    for (int i = 0; i < 4; ++i) {
      const int off = 1 << i;
#pragma unroll
      for (int r = 0; r < 8; ++r) ts[r] += __shfl_xor(ts[r], off, 32);
    }
#pragma unroll
    for (int r = 0; r < 8; ++r) lrow[r] = lrow[r] * alpha[r] + ts[r];
#pragma unroll
    for (int t = 0; t < 8; ++t)
#pragma unroll
      for (int r = 0; r < 8; ++r) acc[t][r] *= alpha[r];

    // P: D-layout f32 -> LDS f16 tile -> reload in A-fragment layout (16x32)
#pragma unroll
    for (int r = 0; r < 8; ++r) {
      lds_p[(r + 8 * half) * 32 + nn] = (h16)s0[r];
      lds_p[(r + 8 * half) * 32 + 16 + nn] = (h16)s1[r];
    }
    __syncthreads();                          // single-wave WG: just the dscnt waits
    v16h pf = load_a_frag(lds_p, 32, 0, 0, lane);
    __syncthreads();

    // Y += P(16x32) * V(32keys x 128d), V fragments already resident
#pragma unroll
    for (int t = 0; t < 8; ++t) acc[t] = wmma_f16(pf, vf[t], acc[t]);
  }

  // normalize + store Y (f16) as [B*T, C] with column h*128 + d
#pragma unroll
  for (int t = 0; t < 8; ++t) {
#pragma unroll
    for (int r = 0; r < 8; ++r) {
      const int m = q0 + r + 8 * half;
      const int col = h * D_ + t * 16 + nn;
      Yb[((size_t)b * T_ + m) * C_ + col] = (h16)(acc[t][r] / lrow[r]);
    }
  }
}

// ---------------- Output projection: out = Y @ Wo^T + bo (f32 out) ----------------
__global__ __launch_bounds__(256) void gemm_o_kernel(const h16* __restrict__ Yh, const h16* __restrict__ Wh,
                                                     const float* __restrict__ bias, float* __restrict__ out) {
  const int lane = threadIdx.x & 31;
  const int gw = (blockIdx.x * blockDim.x + threadIdx.x) >> 5;
  const int ntile = C_ / 64;
  const int m0 = (gw / ntile) * 16;
  const int n0 = (gw % ntile) * 64;
  v8f acc[4];
#pragma unroll
  for (int t = 0; t < 4; ++t)
#pragma unroll
    for (int r = 0; r < 8; ++r) acc[t][r] = 0.f;
  gemm_core(Yh, Wh, C_, m0, n0, lane, acc);
  const int half = lane >> 4, nn = lane & 15;
#pragma unroll
  for (int t = 0; t < 4; ++t) {
    const int n = n0 + 16 * t + nn;
    const float bn = bias[n];
#pragma unroll
    for (int r = 0; r < 8; ++r) {
      const int m = m0 + r + 8 * half;
      out[(size_t)m * C_ + n] = acc[t][r] + bn;
    }
  }
}

extern "C" void kernel_launch(void* const* d_in, const int* in_sizes, int n_in,
                              void* d_out, int out_size, void* d_ws, size_t ws_size,
                              hipStream_t stream) {
  (void)in_sizes; (void)n_in; (void)out_size; (void)ws_size;
  const float* x   = (const float*)d_in[0];
  const float* Wq  = (const float*)d_in[1];
  const float* bq  = (const float*)d_in[2];
  const float* Wkv = (const float*)d_in[3];
  const float* bkv = (const float*)d_in[4];
  const float* Wo  = (const float*)d_in[5];
  const float* bo  = (const float*)d_in[6];
  float* out = (float*)d_out;

  char* ws = (char*)d_ws;
  size_t off = 0;
  auto take = [&](size_t elems) {
    h16* p = (h16*)(ws + off);
    off += (elems * sizeof(h16) + 255) & ~(size_t)255;
    return p;
  };
  h16* xh   = take((size_t)B_ * T_ * C_);        // 32 MB
  h16* Wqh  = take((size_t)C_ * C_);             // 8 MB
  h16* Wkvh = take((size_t)NKV_ * C_);           // 4 MB
  h16* Woh  = take((size_t)C_ * C_);             // 8 MB
  h16* Qh   = take((size_t)B_ * T_ * C_);        // 32 MB
  h16* Kh   = take((size_t)B_ * KVH_ * T_ * D_); // 8 MB
  h16* Vth  = take((size_t)B_ * KVH_ * T_ * D_); // 8 MB
  h16* Yh   = xh;  // x is dead after the projections; reuse its slot for Y

  cvt_kernel<<<4096, 256, 0, stream>>>((const float4*)x,   (v4h*)xh,   B_ * T_ * C_ / 4);
  cvt_kernel<<<2048, 256, 0, stream>>>((const float4*)Wq,  (v4h*)Wqh,  C_ * C_ / 4);
  cvt_kernel<<<1024, 256, 0, stream>>>((const float4*)Wkv, (v4h*)Wkvh, NKV_ * C_ / 4);
  cvt_kernel<<<2048, 256, 0, stream>>>((const float4*)Wo,  (v4h*)Woh,  C_ * C_ / 4);

  // waves = (M/16) * (N/64); threads = waves*32
  gemm_q_kernel<<<(B_ * T_ / 16) * (C_ / 64) / 8, 256, 0, stream>>>(xh, Wqh, bq, Qh);
  gemm_kv_kernel<<<(B_ * T_ / 16) * (NKV_ / 64) / 8, 256, 0, stream>>>(xh, Wkvh, bkv, Kh, Vth);
  attn_kernel<<<B_ * H_ * (T_ / 16), 32, 0, stream>>>(Qh, Kh, Vth, Yh);
  gemm_o_kernel<<<(B_ * T_ / 16) * (C_ / 64) / 8, 256, 0, stream>>>(Yh, Woh, bo, out);
}
